// GraphTripleConv_13941463842944
// MI455X (gfx1250) — matrix operand order repaired and checked by reading, not compiled, not run on hardware
//
#include <hip/hip_runtime.h>
#include <hip/hip_bf16.h>

// Problem constants (match reference)
#define BDIM 32
#define ODIM 4096
#define TDIM 16384
#define HDIM 128
// 1/sqrt(1 + 0.001)
#define RSQ 0.99950037f

typedef __attribute__((ext_vector_type(16))) __bf16 v16bf;
typedef __attribute__((ext_vector_type(8)))  float  v8f;

#define WMMA_BF16(Cacc, Af, Bf) \
  Cacc = __builtin_amdgcn_wmma_f32_16x16x32_bf16(false, Af, false, Bf, (short)0, Cacc, false, false)

// ---- fragment loaders -------------------------------------------------------
// A fragment (16x32, bf16) from row-major source, leading dim ld.
// Lane L in [0,16): row M=L, K = {0..7} U {16..23}; lane L in [16,32): row M=L-16,
// K = {8..15} U {24..31}.  (ISA 7.12.2, 16-bit A 16x32)
__device__ __forceinline__ v16bf ldsA(const __bf16* base, int ld, int lane) {
  const __bf16* p = base + (size_t)(lane & 15) * ld + ((lane >> 4) * 8);
  v16bf a;
#pragma unroll
  for (int e = 0; e < 8; ++e) a[e] = p[e];
#pragma unroll
  for (int e = 0; e < 8; ++e) a[8 + e] = p[16 + e];
  return a;
}

// B fragment (32x16, bf16) from N-major (transposed) weight: baseT points at
// (n0,k0); row n = output column, leading dim ld (= K of the full weight).
// Lane L in [0,16): N=L, K = {0..7} U {16..23}; lane L in [16,32): N=L-16,
// K = {8..15} U {24..31}.
__device__ __forceinline__ v16bf ldsB(const __bf16* baseT, int ld, int lane) {
  const __bf16* p = baseT + (size_t)(lane & 15) * ld + ((lane >> 4) * 8);
  v16bf b;
#pragma unroll
  for (int e = 0; e < 8; ++e) b[e] = p[e];
#pragma unroll
  for (int e = 0; e < 8; ++e) b[8 + e] = p[16 + e];
  return b;
}

// ---- prep: transpose/convert weights to bf16 (N-major), zero pool/counts ----
__global__ void prep_kernel(const float* __restrict__ W1, const float* __restrict__ W2,
                            const float* __restrict__ W3, const float* __restrict__ W4,
                            float* __restrict__ pooled, float* __restrict__ counts,
                            __bf16* __restrict__ w1t, __bf16* __restrict__ w2t,
                            __bf16* __restrict__ w3t, __bf16* __restrict__ w4t) {
  const int i = blockIdx.x * blockDim.x + threadIdx.x;
  if (i < BDIM * ODIM * HDIM) pooled[i] = 0.0f;
  if (i < BDIM * ODIM)        counts[i] = 0.0f;
  if (i < 128 * 384) {                       // W1t[n][k] = W1[k][n], (384,128)
    int n = i / 384, k = i % 384;
    w1t[i] = (__bf16)W1[k * 128 + n];
  }
  if (i < 256 * 128) {                       // W2t[n][k] = W2[k][128+n], (128,384)
    int n = i / 128, k = i % 128;
    w2t[i] = (__bf16)W2[k * 384 + 128 + n];
  }
  if (i < 128 * 128) {                       // W3t / W4t transposed
    int n = i / 128, k = i % 128;
    w3t[i] = (__bf16)W3[k * 128 + n];
    w4t[i] = (__bf16)W4[k * 128 + n];
  }
}

// ---- edge MLP: gather -> GEMM1(384->128) -> GEMM2(128->256) -> write/scatter
__global__ void __launch_bounds__(256, 1)
edge_mlp_kernel(const float* __restrict__ obj,
                const float* __restrict__ pred,
                const int*   __restrict__ edges,
                const __bf16* __restrict__ w1t,   // contiguous with w2t
                const float* __restrict__ b1, const float* __restrict__ g1,
                const float* __restrict__ be1,
                const float* __restrict__ b2, const float* __restrict__ g2,
                const float* __restrict__ be2,
                float* __restrict__ out_p,
                float* __restrict__ pooled,
                float* __restrict__ counts) {
  extern __shared__ char smemRaw[];
  __bf16* sW1 = (__bf16*)smemRaw;            // 128*384 bf16  (96 KB)
  __bf16* sW2 = sW1 + 128 * 384;             // 256*128 bf16  (64 KB)
  __bf16* sH  = sW2 + 256 * 128;             // 8 waves * 16*128 (32 KB)
  int*    sOi = (int*)(sH + 8 * 16 * 128);   // 8 waves * 16 object indices

  const int tid  = threadIdx.x;
  const int lane = tid & 31;
  const int wave = tid >> 5;

  // cooperative weight stage: (49152 + 32768) bf16 = 10240 uint4, 40 per thread
  {
    const uint4* src = (const uint4*)w1t;
    uint4* dst = (uint4*)sW1;
#pragma unroll
    for (int i = 0; i < 40; ++i) dst[tid + 256 * i] = src[tid + 256 * i];
  }
  __syncthreads();

  const int rowBase = blockIdx.x * 128 + wave * 16;     // global edge row base
  const int b   = rowBase / TDIM;                       // batch (constant per block)
  const int m   = lane & 15;
  const int row = rowBase + m;                          // this lane's source row
  const int sidx = edges[2 * row + 0];
  const int oidx = edges[2 * row + 1];
  const float* srcS = obj  + ((size_t)b * ODIM + sidx) * 128;
  const float* srcP = pred + (size_t)row * 128;
  const float* srcO = obj  + ((size_t)b * ODIM + oidx) * 128;

  if (lane < 16) {
    sOi[wave * 16 + lane] = oidx;
    atomicAdd(counts + (size_t)b * ODIM + oidx, 1.0f);
  }

  const int h8 = (lane >> 4) * 8;
  const int rb = (lane >> 4) * 8;
  __bf16* sHw = sH + wave * (16 * 128);

  // ---- GEMM1: x(16x384) @ W1(384x128), A gathered from global, bf16 convert
  v8f c[8] = {};
#pragma unroll
  for (int kc = 0; kc < 12; ++kc) {
    const int kk = kc * 32;
    const float* s = (kk < 128) ? (srcS + kk)
                   : ((kk < 256) ? (srcP + (kk - 128)) : (srcO + (kk - 256)));
    v16bf a;
#pragma unroll
    for (int e = 0; e < 8; ++e) a[e]     = (__bf16)s[h8 + e];
#pragma unroll
    for (int e = 0; e < 8; ++e) a[8 + e] = (__bf16)s[16 + h8 + e];
    // batch B fragments in groups of 4: overlaps DS latency with WMMAs while
    // keeping VGPR pressure below the spill threshold
#pragma unroll
    for (int g = 0; g < 2; ++g) {
      v16bf bm[4];
#pragma unroll
      for (int j = 0; j < 4; ++j)
        bm[j] = ldsB(sW1 + (size_t)((g * 4 + j) * 16) * 384 + kk, 384, lane);
#pragma unroll
      for (int j = 0; j < 4; ++j) WMMA_BF16(c[g * 4 + j], a, bm[j]);
    }
  }
  // epilogue 1: h = relu((x@W1 + b1)*g1/sqrt(1.001) + be1) -> LDS bf16
#pragma unroll
  for (int nt = 0; nt < 8; ++nt) {
    const int col = nt * 16 + (lane & 15);
    const float al = g1[col] * RSQ;
    const float bt = b1[col] * al + be1[col];
#pragma unroll
    for (int r = 0; r < 8; ++r) {
      float v = c[nt][r] * al + bt;
      v = v > 0.f ? v : 0.f;
      sHw[(rb + r) * 128 + col] = (__bf16)v;
    }
  }

  // preload the 4 h-tile A fragments once; reused by both GEMM2 halves
  v16bf ah[4];
#pragma unroll
  for (int kc = 0; kc < 4; ++kc) ah[kc] = ldsA(sHw + kc * 32, 128, lane);

  // hoist per-row object indices for the scatter epilogue
  int oiReg[8];
#pragma unroll
  for (int r = 0; r < 8; ++r) oiReg[r] = sOi[wave * 16 + rb + r];

  // ---- GEMM2 half 0: columns 128..255 of W2 -> new_p (written directly)
  {
    v8f d[8] = {};
#pragma unroll
    for (int kc = 0; kc < 4; ++kc) {
      const int kk = kc * 32;
#pragma unroll
      for (int g = 0; g < 2; ++g) {
        v16bf bm[4];
#pragma unroll
        for (int j = 0; j < 4; ++j)
          bm[j] = ldsB(sW2 + (size_t)((g * 4 + j) * 16) * 128 + kk, 128, lane);
#pragma unroll
        for (int j = 0; j < 4; ++j) WMMA_BF16(d[g * 4 + j], ah[kc], bm[j]);
      }
    }
#pragma unroll
    for (int nt = 0; nt < 8; ++nt) {
      const int col = nt * 16 + (lane & 15);
      const int oc  = 128 + col;
      const float al = g2[oc] * RSQ;
      const float bt = b2[oc] * al + be2[oc];
#pragma unroll
      for (int r = 0; r < 8; ++r) {
        float v = d[nt][r] * al + bt;
        v = v > 0.f ? v : 0.f;
        out_p[(size_t)(rowBase + rb + r) * 128 + col] = v;
      }
    }
  }

  // ---- GEMM2 half 1: columns 256..383 of W2 -> new_o, scatter-add into pool
  {
    v8f d[8] = {};
#pragma unroll
    for (int kc = 0; kc < 4; ++kc) {
      const int kk = kc * 32;
#pragma unroll
      for (int g = 0; g < 2; ++g) {
        v16bf bm[4];
#pragma unroll
        for (int j = 0; j < 4; ++j)
          bm[j] = ldsB(sW2 + (size_t)(128 + (g * 4 + j) * 16) * 128 + kk, 128, lane);
#pragma unroll
        for (int j = 0; j < 4; ++j) WMMA_BF16(d[g * 4 + j], ah[kc], bm[j]);
      }
    }
#pragma unroll
    for (int nt = 0; nt < 8; ++nt) {
      const int col = nt * 16 + (lane & 15);
      const int oc  = 256 + col;
      const float al = g2[oc] * RSQ;
      const float bt = b2[oc] * al + be2[oc];
#pragma unroll
      for (int r = 0; r < 8; ++r) {
        float v = d[nt][r] * al + bt;
        v = v > 0.f ? v : 0.f;
        atomicAdd(pooled + ((size_t)b * ODIM + oiReg[r]) * 128 + col, v);
      }
    }
  }
}

// ---- object MLP: normalize pooled, GEMM3 -> GEMM4 -> new_obj ---------------
__global__ void __launch_bounds__(256, 1)
obj_mlp_kernel(const float* __restrict__ pooled,
               const float* __restrict__ counts,
               const __bf16* __restrict__ w3t,  // contiguous with w4t
               const float* __restrict__ b3, const float* __restrict__ g3,
               const float* __restrict__ be3,
               const float* __restrict__ b4, const float* __restrict__ g4,
               const float* __restrict__ be4,
               float* __restrict__ out_obj) {
  extern __shared__ char smemRaw[];
  __bf16* sW3 = (__bf16*)smemRaw;            // 128*128
  __bf16* sW4 = sW3 + 128 * 128;             // 128*128
  __bf16* sH  = sW4 + 128 * 128;             // 8 * 16*128

  const int tid  = threadIdx.x;
  const int lane = tid & 31;
  const int wave = tid >> 5;

  { // stage W3t+W4t: 32768 bf16 = 4096 uint4, 16 per thread
    const uint4* src = (const uint4*)w3t;
    uint4* dst = (uint4*)sW3;
#pragma unroll
    for (int i = 0; i < 16; ++i) dst[tid + 256 * i] = src[tid + 256 * i];
  }
  __syncthreads();

  const int rowBase = blockIdx.x * 128 + wave * 16;   // in [0, B*O)
  const int row = rowBase + (lane & 15);
  const float cnt = counts[row];
  const float inv = 1.0f / (cnt > 1.0f ? cnt : 1.0f);
  const float* srcX = pooled + (size_t)row * 128;
  const int h8 = (lane >> 4) * 8;
  const int rb = (lane >> 4) * 8;
  __bf16* sHw = sH + wave * (16 * 128);

  // GEMM3: (pooled/cnt)(16x128) @ W3(128x128)
  v8f c[8] = {};
#pragma unroll
  for (int kc = 0; kc < 4; ++kc) {
    const int kk = kc * 32;
    v16bf a;
#pragma unroll
    for (int e = 0; e < 8; ++e) a[e]     = (__bf16)(srcX[kk + h8 + e] * inv);
#pragma unroll
    for (int e = 0; e < 8; ++e) a[8 + e] = (__bf16)(srcX[kk + 16 + h8 + e] * inv);
#pragma unroll
    for (int g = 0; g < 2; ++g) {
      v16bf bm[4];
#pragma unroll
      for (int j = 0; j < 4; ++j)
        bm[j] = ldsB(sW3 + (size_t)((g * 4 + j) * 16) * 128 + kk, 128, lane);
#pragma unroll
      for (int j = 0; j < 4; ++j) WMMA_BF16(c[g * 4 + j], a, bm[j]);
    }
  }
#pragma unroll
  for (int nt = 0; nt < 8; ++nt) {
    const int col = nt * 16 + (lane & 15);
    const float al = g3[col] * RSQ;
    const float bt = b3[col] * al + be3[col];
#pragma unroll
    for (int r = 0; r < 8; ++r) {
      float v = c[nt][r] * al + bt;
      v = v > 0.f ? v : 0.f;
      sHw[(rb + r) * 128 + col] = (__bf16)v;
    }
  }

  // GEMM4: h(16x128) @ W4(128x128) -> new_obj
  v16bf ah[4];
#pragma unroll
  for (int kc = 0; kc < 4; ++kc) ah[kc] = ldsA(sHw + kc * 32, 128, lane);

  v8f d[8] = {};
#pragma unroll
  for (int kc = 0; kc < 4; ++kc) {
    const int kk = kc * 32;
#pragma unroll
    for (int g = 0; g < 2; ++g) {
      v16bf bm[4];
#pragma unroll
      for (int j = 0; j < 4; ++j)
        bm[j] = ldsB(sW4 + (size_t)((g * 4 + j) * 16) * 128 + kk, 128, lane);
#pragma unroll
      for (int j = 0; j < 4; ++j) WMMA_BF16(d[g * 4 + j], ah[kc], bm[j]);
    }
  }
#pragma unroll
  for (int nt = 0; nt < 8; ++nt) {
    const int col = nt * 16 + (lane & 15);
    const float al = g4[col] * RSQ;
    const float bt = b4[col] * al + be4[col];
#pragma unroll
    for (int r = 0; r < 8; ++r) {
      float v = d[nt][r] * al + bt;
      v = v > 0.f ? v : 0.f;
      out_obj[(size_t)(rowBase + rb + r) * 128 + col] = v;
    }
  }
}

extern "C" void kernel_launch(void* const* d_in, const int* in_sizes, int n_in,
                              void* d_out, int out_size, void* d_ws, size_t ws_size,
                              hipStream_t stream) {
  (void)in_sizes; (void)n_in; (void)out_size; (void)ws_size;
  const float* obj   = (const float*)d_in[0];
  const float* pred  = (const float*)d_in[1];
  const int*   edges = (const int*)d_in[2];
  const float* W1  = (const float*)d_in[3];
  const float* b1  = (const float*)d_in[4];
  const float* g1  = (const float*)d_in[5];
  const float* be1 = (const float*)d_in[6];
  const float* W2  = (const float*)d_in[7];
  const float* b2  = (const float*)d_in[8];
  const float* g2  = (const float*)d_in[9];
  const float* be2 = (const float*)d_in[10];
  const float* W3  = (const float*)d_in[11];
  const float* b3  = (const float*)d_in[12];
  const float* g3  = (const float*)d_in[13];
  const float* be3 = (const float*)d_in[14];
  const float* W4  = (const float*)d_in[15];
  const float* b4  = (const float*)d_in[16];
  const float* g4  = (const float*)d_in[17];
  const float* be4 = (const float*)d_in[18];

  // workspace layout
  float*  pooled = (float*)d_ws;                              // B*O*128 f32
  float*  counts = pooled + (size_t)BDIM * ODIM * HDIM;       // B*O f32
  __bf16* w1t = (__bf16*)(counts + (size_t)BDIM * ODIM);      // 128*384
  __bf16* w2t = w1t + 128 * 384;                              // 256*128
  __bf16* w3t = w2t + 256 * 128;                              // 128*128
  __bf16* w4t = w3t + 128 * 128;                              // 128*128

  float* out_obj = (float*)d_out;                             // B*O*128
  float* out_p   = out_obj + (size_t)BDIM * ODIM * HDIM;      // B*T*128

  const int smemEdge = (128 * 384 + 256 * 128 + 8 * 16 * 128) * 2 + 8 * 16 * 4;
  const int smemObj  = (128 * 128 + 128 * 128 + 8 * 16 * 128) * 2;
  hipFuncSetAttribute((const void*)edge_mlp_kernel,
                      hipFuncAttributeMaxDynamicSharedMemorySize, smemEdge);
  hipFuncSetAttribute((const void*)obj_mlp_kernel,
                      hipFuncAttributeMaxDynamicSharedMemorySize, smemObj);

  // 1) prep: zero pool/counts, bf16-transpose weights
  const int prepN = BDIM * ODIM * HDIM;                       // largest range
  prep_kernel<<<(prepN + 255) / 256, 256, 0, stream>>>(
      W1, W2, W3, W4, pooled, counts, w1t, w2t, w3t, w4t);

  // 2) edge MLP over B*T = 524288 rows, 128 rows per block
  edge_mlp_kernel<<<(BDIM * TDIM) / 128, 256, smemEdge, stream>>>(
      obj, pred, edges, w1t, b1, g1, be1, b2, g2, be2, out_p, pooled, counts);

  // 3) object MLP over B*O = 131072 rows, 128 rows per block
  obj_mlp_kernel<<<(BDIM * ODIM) / 128, 256, smemObj, stream>>>(
      pooled, counts, w3t, b3, g3, be3, b4, g4, be4, out_obj);
}
